// GeometricPlainGnn_29618094473563
// MI455X (gfx1250) — matrix operand order repaired and checked by reading, not compiled, not run on hardware
//
#include <hip/hip_runtime.h>

typedef _Float16 f16;
typedef __attribute__((ext_vector_type(16))) _Float16 v16h;
typedef __attribute__((ext_vector_type(8)))  float    v8f;

#define C_DIM   96
#define NNODES  50000
#define NEDGES  800000

struct f8 { float v[8]; };

__device__ __forceinline__ f8 ld8(const float* p) {   // 8 consecutive floats, 32B aligned
    f8 r;
    float4 a = *(const float4*)p;
    float4 b = *(const float4*)(p + 4);
    r.v[0]=a.x; r.v[1]=a.y; r.v[2]=a.z; r.v[3]=a.w;
    r.v[4]=b.x; r.v[5]=b.y; r.v[6]=b.z; r.v[7]=b.w;
    return r;
}

// Fill LDS with W pre-swizzled into WMMA B-fragment order:
// frag f = nt*3+kt; element (f, lane, h) = W[k][n],
//   k = kt*32 + ((lane>>4)<<3) + (h&7) + ((h>>3)<<4),  n = nt*16 + (lane&15)   (ISA 7.12.2 B layout)
// A B fragment is then 32 contiguous bytes per lane -> 2x ds_load_b128.
__device__ __forceinline__ void fill_wfrag(f16* Wf, const float* __restrict__ W, int tstride) {
    for (int i = threadIdx.x; i < C_DIM * C_DIM; i += tstride) {
        const int f    = i >> 9;          // /512
        const int rem  = i & 511;
        const int lane = rem >> 4;
        const int h    = rem & 15;
        const int nt = f / 3, kt = f % 3;
        const int k = kt * 32 + ((lane >> 4) << 3) + (h & 7) + ((h >> 3) << 4);
        const int n = nt * 16 + (lane & 15);
        Wf[i] = (f16)W[k * C_DIM + n];
    }
}

__device__ __forceinline__ v16h load_bfrag(const f16* Wf, int kt, int nt, int lane) {
    return *(const v16h*)(Wf + (nt * 3 + kt) * 512 + lane * 16);   // 32B contiguous
}

__device__ __forceinline__ v8f bcast8(float b) {
    v8f c;
    #pragma unroll
    for (int i = 0; i < 8; ++i) c[i] = b;
    return c;
}

// ---------------------------------------------------------------------------
// Fused GINE message+aggregate, e never materialized:
//   e_tile = relu(a * W_l0 + b_l0) @ W_l1 + b_l1         (WMMA, bias in C-init)
//   agg[dst] += relu(x[src] + e_tile)                    (L2-resident atomics)
// One wave per 16-edge tile, 8 waves/block. E/16 = 50000 tiles = 6250*8.
// ---------------------------------------------------------------------------
__global__ __launch_bounds__(256) void edge_conv_kernel(
    const float* __restrict__ xin, const float* __restrict__ edge_attr,
    const int* __restrict__ src, const int* __restrict__ dst,
    const float* __restrict__ W0, const float* __restrict__ b0,
    const float* __restrict__ W1, const float* __restrict__ b1,
    float* __restrict__ agg)
{
    __shared__ alignas(16) f16   Wf[C_DIM * C_DIM];
    __shared__ alignas(16) float w0l[C_DIM], b0l[C_DIM], b1l[C_DIM];

    const int t = threadIdx.x;
    fill_wfrag(Wf, W1, 256);
    if (t < C_DIM) { w0l[t] = W0[t]; b0l[t] = b0[t]; b1l[t] = b1[t]; }
    __syncthreads();

    const int wave = t >> 5, lane = t & 31;
    const int tile = blockIdx.x * 8 + wave;
    const int e0   = tile * 16;

    const float av = edge_attr[e0 + (lane & 15)];   // this lane's edge scalar (row M)

    // hidden(m,k) = relu(a_m * W0[k] + b0[k]); A layout = two runs of 8 consecutive K
    v16h aa[3];
    #pragma unroll
    for (int kt = 0; kt < 3; ++kt) {
        const int kb = kt * 32 + ((lane >> 4) << 3);
        const f8 wlo = ld8(w0l + kb), whi = ld8(w0l + kb + 16);
        const f8 blo = ld8(b0l + kb), bhi = ld8(b0l + kb + 16);
        #pragma unroll
        for (int j = 0; j < 8; ++j) {
            aa[kt][j]     = (f16)fmaxf(av * wlo.v[j] + blo.v[j], 0.0f);
            aa[kt][8 + j] = (f16)fmaxf(av * whi.v[j] + bhi.v[j], 0.0f);
        }
    }

    const int col   = lane & 15;
    const int rbase = e0 + ((lane >> 4) << 3);

    v8f acc[6];
    #pragma unroll
    for (int nt = 0; nt < 6; ++nt) {
        v8f c = bcast8(b1l[nt * 16 + col]);        // bias folded into accumulator
        #pragma unroll
        for (int kt = 0; kt < 3; ++kt) {
            const v16h bb = load_bfrag(Wf, kt, nt, lane);
            c = __builtin_amdgcn_wmma_f32_16x16x32_f16(
                false, aa[kt], false, bb, (short)0, c, false, false);
        }
        acc[nt] = c;
    }

    // gather + message + scatter-add; each (r,nt) is a coalesced 64B segment
    #pragma unroll
    for (int r = 0; r < 8; ++r) {
        const int edge = rbase + r;
        const int s = src[edge], d = dst[edge];
        const float* xr = xin + (size_t)s * C_DIM;
        float*       ar = agg + (size_t)d * C_DIM;
        #pragma unroll
        for (int nt = 0; nt < 6; ++nt) {
            const float msg = fmaxf(xr[nt * 16 + col] + acc[nt][r], 0.0f);
            atomicAdd(ar + nt * 16 + col, msg);
        }
    }
}

// ---------------------------------------------------------------------------
__global__ void zero_kernel(float* __restrict__ p, int n)
{
    int i = blockIdx.x * blockDim.x + threadIdx.x;
    if (i < n) p[i] = 0.0f;
}

// ---------------------------------------------------------------------------
// out = relu( relu( (x+agg) @ W1 + b1 ) @ W2 + b2 )
// One wave per 16-node tile, 5 waves/block. N/16 = 3125 = 625*5 (no tail).
// ---------------------------------------------------------------------------
__global__ __launch_bounds__(160) void node_mlp_kernel(
    const float* __restrict__ xin, const float* __restrict__ agg,
    const float* __restrict__ W1, const float* __restrict__ b1,
    const float* __restrict__ W2, const float* __restrict__ b2,
    float* __restrict__ out)
{
    __shared__ alignas(16) f16   Wf1[C_DIM * C_DIM];
    __shared__ alignas(16) f16   Wf2[C_DIM * C_DIM];
    __shared__ alignas(16) float b1l[C_DIM], b2l[C_DIM];
    __shared__ alignas(16) f16   U[5][3 * 512];   // per-wave layer-1 output, A-frag order

    const int t = threadIdx.x;
    fill_wfrag(Wf1, W1, 160);
    fill_wfrag(Wf2, W2, 160);
    if (t < C_DIM) { b1l[t] = b1[t]; b2l[t] = b2[t]; }
    __syncthreads();

    const int wave = t >> 5, lane = t & 31;
    const int tile = blockIdx.x * 5 + wave;
    const int row0 = tile * 16;
    f16* Uw = U[wave];

    const int m   = lane & 15, col = lane & 15;
    const int rb  = (lane >> 4) << 3;

    // ---- layer 1: A fragments directly from global (x + agg), aligned float4 runs ----
    v16h aa[3];
    #pragma unroll
    for (int kt = 0; kt < 3; ++kt) {
        const int kb = kt * 32 + ((lane >> 4) << 3);
        const size_t base = (size_t)(row0 + m) * C_DIM + kb;
        const f8 xlo = ld8(xin + base),     xhi = ld8(xin + base + 16);
        const f8 glo = ld8(agg + base),     ghi = ld8(agg + base + 16);
        #pragma unroll
        for (int j = 0; j < 8; ++j) {
            aa[kt][j]     = (f16)(xlo.v[j] + glo.v[j]);
            aa[kt][8 + j] = (f16)(xhi.v[j] + ghi.v[j]);
        }
    }

    v8f acc[6];
    #pragma unroll
    for (int nt = 0; nt < 6; ++nt) {
        v8f c = bcast8(b1l[nt * 16 + col]);        // bias folded into accumulator
        #pragma unroll
        for (int kt = 0; kt < 3; ++kt) {
            const v16h bb = load_bfrag(Wf1, kt, nt, lane);
            c = __builtin_amdgcn_wmma_f32_16x16x32_f16(
                false, aa[kt], false, bb, (short)0, c, false, false);
        }
        acc[nt] = c;
    }

    // relu, scatter into U in layer-2 A-fragment order (transpose via LDS).
    // Element (row = rb+r, k2 = nt*16+col):
    //   kt2 = k2>>5; kk = k2&31; lane_hi = (kk>>3)&1; h2 = ((kk>>4)<<3) + (kk&7)
    #pragma unroll
    for (int nt = 0; nt < 6; ++nt) {
        const int k2      = nt * 16 + col;
        const int kt2     = k2 >> 5;
        const int kk      = k2 & 31;
        const int lane_hi = (kk >> 3) & 1;
        const int h2      = ((kk >> 4) << 3) + (kk & 7);
        #pragma unroll
        for (int r = 0; r < 8; ++r)
            Uw[kt2 * 512 + (lane_hi * 16 + rb + r) * 16 + h2] =
                (f16)fmaxf(acc[nt][r], 0.0f);
    }
    asm volatile("s_wait_dscnt 0" ::: "memory");

    // ---- layer 2: A fragments are contiguous 32B per lane ----
    #pragma unroll
    for (int kt = 0; kt < 3; ++kt)
        aa[kt] = *(const v16h*)(Uw + kt * 512 + lane * 16);

    #pragma unroll
    for (int nt = 0; nt < 6; ++nt) {
        v8f c = bcast8(b2l[nt * 16 + col]);        // bias folded into accumulator
        #pragma unroll
        for (int kt = 0; kt < 3; ++kt) {
            const v16h bb = load_bfrag(Wf2, kt, nt, lane);
            c = __builtin_amdgcn_wmma_f32_16x16x32_f16(
                false, aa[kt], false, bb, (short)0, c, false, false);
        }
        #pragma unroll
        for (int r = 0; r < 8; ++r)
            out[(size_t)(row0 + rb + r) * C_DIM + nt * 16 + col] =
                fmaxf(c[r], 0.0f);                 // final relu (outer relu idempotent)
    }
}

// ---------------------------------------------------------------------------
extern "C" void kernel_launch(void* const* d_in, const int* in_sizes, int n_in,
                              void* d_out, int out_size, void* d_ws, size_t ws_size,
                              hipStream_t stream)
{
    const float* x    = (const float*)d_in[0];
    const int*   ei   = (const int*)  d_in[1];   // [2,E]: row0=src, row1=dst
    const float* ea   = (const float*)d_in[2];   // [E,1]
    const float* W_l0 = (const float*)d_in[3];
    const float* b_l0 = (const float*)d_in[4];
    const float* W_l1 = (const float*)d_in[5];
    const float* b_l1 = (const float*)d_in[6];
    const float* W00  = (const float*)d_in[7];
    const float* b00  = (const float*)d_in[8];
    const float* W01  = (const float*)d_in[9];
    const float* b01  = (const float*)d_in[10];
    const float* W10  = (const float*)d_in[11];
    const float* b10  = (const float*)d_in[12];
    const float* W11  = (const float*)d_in[13];
    const float* b11  = (const float*)d_in[14];

    const int* src = ei;
    const int* dst = ei + NEDGES;

    // workspace layout (floats): agg[N*96] | h1[N*96]   (only 38.4 MB; L2-resident)
    float* agg = (float*)d_ws;
    float* h1  = agg + (size_t)NNODES * C_DIM;
    float* out = (float*)d_out;

    const int naggr   = NNODES * C_DIM;
    const int zblocks = (naggr + 255) / 256;
    const int eblocks = NEDGES / 128;        // 6250
    const int nblocks = (NNODES / 16) / 5;   // 625

    // conv0: fused edge-MLP (WMMA) + gather + scatter-add, then node MLP (WMMA)
    zero_kernel<<<zblocks, 256, 0, stream>>>(agg, naggr);
    edge_conv_kernel<<<eblocks, 256, 0, stream>>>(x, ea, src, dst,
                                                  W_l0, b_l0, W_l1, b_l1, agg);
    node_mlp_kernel<<<nblocks, 160, 0, stream>>>(x, agg, W00, b00, W01, b01, h1);

    // conv1: recompute e on the fly (14.7 GFLOP of WMMA ≪ 307 MB of HBM saved)
    zero_kernel<<<zblocks, 256, 0, stream>>>(agg, naggr);
    edge_conv_kernel<<<eblocks, 256, 0, stream>>>(h1, ea, src, dst,
                                                  W_l0, b_l0, W_l1, b_l1, agg);
    node_mlp_kernel<<<nblocks, 160, 0, stream>>>(h1, agg, W10, b10, W11, b11, out);
}
